// Gate_26422638805112
// MI455X (gfx1250) — matrix operand order repaired and checked by reading, not compiled, not run on hardware
//
#include <hip/hip_runtime.h>
#include <hip/hip_bf16.h>
#include <cstdint>

typedef __bf16 v16bf __attribute__((ext_vector_type(16)));
typedef __bf16 v8bf  __attribute__((ext_vector_type(8)));
typedef float  v8f   __attribute__((ext_vector_type(8)));
typedef float  v4f   __attribute__((ext_vector_type(4)));
typedef int    i32x4 __attribute__((vector_size(16)));   // matches builtin's pointee type

#define T_TOKENS     32768
#define DDIM         7168
#define NEXP         256
#define NGROUP       8
#define GSIZE        32
#define TOPKG        4
#define TOPK         8
#define ROUTE_SCALE  2.5f

#define WAVES         8
#define TOK_PER_WAVE  16
#define TOK_PER_BLOCK 128
#define NTILES        16      // 256 experts / 16
#define KSTEP         32
#define ROWSTRIDE     257     // score row stride (floats): conflict-free scan
#define BROW          40      // B staging row stride (bf16): 80B, conflict-free b128 reads
#define SB_BYTES      (NEXP * BROW * 2)   // 20480 B per K-step buffer

// ---- CDNA5 async global->LDS path (guarded; fallback = load + ds_store) ----
#if defined(__has_builtin)
#if __has_builtin(__builtin_amdgcn_global_load_async_to_lds_b128)
#define HAVE_ASYNC_LDS 1
#endif
#if __has_builtin(__builtin_amdgcn_s_wait_asynccnt)
#define HAVE_WAIT_ASYNC_BUILTIN 1
#endif
#endif

typedef __attribute__((address_space(1))) i32x4 as1_i32x4;
typedef __attribute__((address_space(3))) i32x4 as3_i32x4;

__device__ __forceinline__ void cp16_g2l(const void* g, void* l) {
#if defined(HAVE_ASYNC_LDS)
    __builtin_amdgcn_global_load_async_to_lds_b128(
        (as1_i32x4*)(uintptr_t)g,
        (as3_i32x4*)(unsigned int)(uintptr_t)l,
        0, 0);
#else
    *(uint4*)l = *(const uint4*)g;
#endif
}

__device__ __forceinline__ void wait_async_lds() {
#if defined(HAVE_ASYNC_LDS)
#if defined(HAVE_WAIT_ASYNC_BUILTIN)
    __builtin_amdgcn_s_wait_asynccnt(0);
#else
    asm volatile("s_wait_asynccnt 0x0" ::: "memory");
#endif
#endif
}

// ---------------------------------------------------------------------------
// Kernel 1: convert router weight [E, D] fp32 -> bf16 into workspace.
// ---------------------------------------------------------------------------
__global__ void cvt_w_bf16(const float* __restrict__ W, __bf16* __restrict__ Wb) {
    size_t i = ((size_t)blockIdx.x * blockDim.x + threadIdx.x) * 8;
    v4f a = *(const v4f*)(W + i);
    v4f b = *(const v4f*)(W + i + 4);
    v8bf o;
    o[0] = (__bf16)a[0]; o[1] = (__bf16)a[1]; o[2] = (__bf16)a[2]; o[3] = (__bf16)a[3];
    o[4] = (__bf16)b[0]; o[5] = (__bf16)b[1]; o[6] = (__bf16)b[2]; o[7] = (__bf16)b[3];
    *(v8bf*)(Wb + i) = o;
}

// ---------------------------------------------------------------------------
// Helpers for the main kernel
// ---------------------------------------------------------------------------
// Stage B[k0..k0+32) for all 256 experts into an LDS buffer: thread e copies
// one 64-byte row chunk with four async b128 transfers.
__device__ __forceinline__ void stage_b(const __bf16* __restrict__ Wb,
                                        __bf16* __restrict__ buf,
                                        int e, int k0) {
    const __bf16* g = Wb + (size_t)e * DDIM + k0;
    __bf16* l = buf + e * BROW;
    cp16_g2l(g,      l);
    cp16_g2l(g + 8,  l + 8);
    cp16_g2l(g + 16, l + 16);
    cp16_g2l(g + 24, l + 24);
}

// One K-step: issue A-prefetch for knext, convert current A, then 16 WMMAs
// fed by a 2-deep rolling LDS B-tile pipeline (load t+2 before WMMA t).
__device__ __forceinline__ void compute_step(const float* __restrict__ xrow,
                                             int knext,
                                             const __bf16* __restrict__ bbase,
                                             v8f* __restrict__ acc,
                                             v4f* __restrict__ araw) {
    v4f c0 = araw[0], c1 = araw[1], c2 = araw[2], c3 = araw[3];
    // prefetch next step's raw A while this step computes
    araw[0] = *(const v4f*)(xrow + knext);
    araw[1] = *(const v4f*)(xrow + knext + 4);
    araw[2] = *(const v4f*)(xrow + knext + 8);
    araw[3] = *(const v4f*)(xrow + knext + 12);

    v16bf A;
    A[0]  = (__bf16)c0[0]; A[1]  = (__bf16)c0[1]; A[2]  = (__bf16)c0[2]; A[3]  = (__bf16)c0[3];
    A[4]  = (__bf16)c1[0]; A[5]  = (__bf16)c1[1]; A[6]  = (__bf16)c1[2]; A[7]  = (__bf16)c1[3];
    A[8]  = (__bf16)c2[0]; A[9]  = (__bf16)c2[1]; A[10] = (__bf16)c2[2]; A[11] = (__bf16)c2[3];
    A[12] = (__bf16)c3[0]; A[13] = (__bf16)c3[1]; A[14] = (__bf16)c3[2]; A[15] = (__bf16)c3[3];

    v16bf bcur = *(const v16bf*)(bbase);
    v16bf bnxt = *(const v16bf*)(bbase + 16 * BROW);
#pragma unroll
    for (int t = 0; t < NTILES; ++t) {
        v16bf bpre = bnxt;
        if (t + 2 < NTILES)
            bpre = *(const v16bf*)(bbase + (t + 2) * 16 * BROW);  // 2 tiles ahead
        acc[t] = __builtin_amdgcn_wmma_f32_16x16x32_bf16(
            false, A, false, bcur, (short)0, acc[t], false, false);
        bcur = bnxt;
        bnxt = bpre;
    }
}

// ---------------------------------------------------------------------------
// Kernel 2: scores = sigmoid(x @ W^T) via WMMA bf16 with LDS-staged B,
// then DeepSeek-style grouped top-k routing per token.
// ---------------------------------------------------------------------------
__global__ __launch_bounds__(256, 1) void moe_gate(
    const float* __restrict__ x, const __bf16* __restrict__ Wb,
    const float* __restrict__ bias, float* __restrict__ w_out,
    int* __restrict__ idx_out)
{
    extern __shared__ char smem[];
    float*  sBias   = (float*)smem;                          // [256]   1 KB
    __bf16* sB0     = (__bf16*)(smem + 1024);                // 20480 B
    __bf16* sB1     = (__bf16*)(smem + 1024 + SB_BYTES);     // 20480 B
    float*  sScores = (float*)(smem + 1024);                 // 131584 B (reuses sB0/sB1)

    const int tid  = threadIdx.x;
    const int wave = tid >> 5;
    const int lane = tid & 31;
    const int col  = lane & 15;                  // A row (token) / B col (expert)
    const int half = lane >> 4;                  // which K-half this lane carries

    sBias[tid] = bias[tid];                      // 256 threads, 256 experts

    const int tokBase = blockIdx.x * TOK_PER_BLOCK + wave * TOK_PER_WAVE;
    const int myTok   = tokBase + col;

    // A: elements 0..15 at K = k0 + 16*half .. +15
    const float*  xrow  = x + (size_t)myTok * DDIM + half * 16;
    // B from LDS: lane holds K = k0 + 16*half .. +15 for expert (t*16 + col)
    const __bf16* bb0 = sB0 + col * BROW + half * 16;
    const __bf16* bb1 = sB1 + col * BROW + half * 16;

    v8f acc[NTILES];
    v8f zero = {0.f, 0.f, 0.f, 0.f, 0.f, 0.f, 0.f, 0.f};
#pragma unroll
    for (int t = 0; t < NTILES; ++t) acc[t] = zero;

    // Preload A(0) and stage B(0) into sB0.
    v4f araw[4];
    araw[0] = *(const v4f*)(xrow);
    araw[1] = *(const v4f*)(xrow + 4);
    araw[2] = *(const v4f*)(xrow + 8);
    araw[3] = *(const v4f*)(xrow + 12);
    stage_b(Wb, sB0, tid, 0);
    wait_async_lds();
    __syncthreads();

    // Unroll-by-2 over K with ping-pong LDS buffers; stage next buffer
    // BEFORE computing on the current one so the async copy overlaps compute.
    for (int k0 = 0; k0 < DDIM; k0 += 2 * KSTEP) {
        stage_b(Wb, sB1, tid, k0 + KSTEP);                  // k0+32 <= 7136 always valid
        compute_step(xrow, k0 + KSTEP, bb0, acc, araw);
        wait_async_lds();
        __syncthreads();

        int k2 = k0 + 2 * KSTEP;
        if (k2 < DDIM) stage_b(Wb, sB0, tid, k2);
        compute_step(xrow, (k2 < DDIM) ? k2 : 0, bb1, acc, araw);
        wait_async_lds();
        __syncthreads();
    }

    // Epilogue: sigmoid and scatter to LDS. C layout: VGPR r holds row
    // M = r + 8*half, lane col = expert column within the tile.
#pragma unroll
    for (int t = 0; t < NTILES; ++t) {
#pragma unroll
        for (int r = 0; r < 8; ++r) {
            int tl = wave * TOK_PER_WAVE + half * 8 + r;
            int e  = t * 16 + col;
            float v = acc[t][r];
            sScores[tl * ROWSTRIDE + e] = 1.0f / (1.0f + __expf(-v));
        }
    }
    __syncthreads();

    // Per-token grouped routing: one token per thread (threads 0..127).
    if (tid < TOK_PER_BLOCK) {
        const float* sc = sScores + tid * ROWSTRIDE;

        // group score = sum of top-2 of (score + bias) in each group of 32
        float gs[NGROUP];
#pragma unroll
        for (int g = 0; g < NGROUP; ++g) {
            float m1 = -1e30f, m2 = -1e30f;
            for (int j = 0; j < GSIZE; ++j) {
                float v = sc[g * GSIZE + j] + sBias[g * GSIZE + j];
                if (v > m1) { m2 = m1; m1 = v; }
                else if (v > m2) { m2 = v; }
            }
            gs[g] = m1 + m2;
        }
        // top-4 groups (strict > keeps lowest index on ties, like lax.top_k)
        unsigned gmask = 0;
#pragma unroll
        for (int p = 0; p < TOPKG; ++p) {
            int best = 0; float bv = -1e30f;
            for (int g = 0; g < NGROUP; ++g)
                if (!((gmask >> g) & 1u) && gs[g] > bv) { bv = gs[g]; best = g; }
            gmask |= 1u << best;
        }
        // top-8 experts among the selected groups, ranked by (score + bias)
        unsigned chosen[NGROUP] = {0, 0, 0, 0, 0, 0, 0, 0};
        int   ids[TOPK];
        float wts[TOPK];
        float sum = 0.f;
#pragma unroll
        for (int p = 0; p < TOPK; ++p) {
            int best = 0; float bv = -1e30f;
            for (int g = 0; g < NGROUP; ++g) {
                if (!((gmask >> g) & 1u)) continue;
                unsigned cm = chosen[g];
                for (int j = 0; j < GSIZE; ++j) {
                    if ((cm >> j) & 1u) continue;
                    int e = g * GSIZE + j;
                    float v = sc[e] + sBias[e];
                    if (v > bv) { bv = v; best = e; }
                }
            }
            chosen[best >> 5] |= 1u << (best & 31);
            ids[p] = best;
            float w = sc[best];          // ORIGINAL sigmoid score (no bias)
            wts[p] = w;
            sum += w;
        }
        float inv = ROUTE_SCALE / sum;
        int tok = blockIdx.x * TOK_PER_BLOCK + tid;
#pragma unroll
        for (int p = 0; p < TOPK; ++p) {
            w_out[tok * TOPK + p]   = wts[p] * inv;
            idx_out[tok * TOPK + p] = ids[p];
        }
    }
}

// ---------------------------------------------------------------------------
extern "C" void kernel_launch(void* const* d_in, const int* in_sizes, int n_in,
                              void* d_out, int out_size, void* d_ws, size_t ws_size,
                              hipStream_t stream) {
    const float* x    = (const float*)d_in[0];   // [T, D] fp32
    const float* W    = (const float*)d_in[1];   // [E, D] fp32
    const float* bias = (const float*)d_in[2];   // [E]    fp32

    __bf16* Wb = (__bf16*)d_ws;                  // [E, D] bf16 (3.67 MB of scratch)

    float* w_out   = (float*)d_out;                                    // [T, 8] weights
    int*   idx_out = (int*)((float*)d_out + (size_t)T_TOKENS * TOPK);  // [T, 8] indices

    // W fp32 -> bf16: E*D/8 = 229376 threads
    cvt_w_bf16<<<(NEXP * DDIM / 8) / 256, 256, 0, stream>>>(W, Wb);

    size_t smem = 1024 + (size_t)TOK_PER_BLOCK * ROWSTRIDE * sizeof(float); // 132608 B
    moe_gate<<<T_TOKENS / TOK_PER_BLOCK, 256, smem, stream>>>(x, Wb, bias, w_out, idx_out);
}
